// TwoBranchEncoderLayer_25744033972255
// MI455X (gfx1250) — matrix operand order repaired and checked by reading, not compiled, not run on hardware
//
#include <hip/hip_runtime.h>
#include <math.h>

// ---------------------------------------------------------------------------
// TwoBranchEncoderLayer for MI455X (gfx1250, wave32, WMMA bf16, async-to-LDS)
// ---------------------------------------------------------------------------

typedef __attribute__((ext_vector_type(16))) __bf16 v16bf;
typedef __attribute__((ext_vector_type(8)))  float  v8f;

#define Bsz   4
#define Nseq  512
#define Dm    2048
#define Ff    8192
#define NH    16
#define HD    128
#define FILMD 64
#define MROWS (Bsz * Nseq)      // 2048 token rows

// GEMM tiling: block tile 128x128, K-step 32, 8 waves of 32 (wave tile 32x64)
#define BM 128
#define BN 128
#define BK 32
#define LSA 40   // padded LDS row stride (elements) for A tile [m][k]
#define LSB 40   // padded LDS row stride (elements) for B tile [n][k]

union Frag  { uint4 q[2]; v16bf v; };
union Acc8  { v8f v; float f[8]; };

__device__ __forceinline__ unsigned short f2bf(float f) {
  unsigned int u = __float_as_uint(f);
  u += 0x7FFFu + ((u >> 16) & 1u);            // round-to-nearest-even
  return (unsigned short)(u >> 16);
}

// Async global->LDS 16-byte copy (CDNA5 GLOBAL_LOAD_ASYNC_TO_LDS_B128,
// tracked by ASYNCcnt).  ldsOff = wave-relative LDS byte address (low 32 bits
// of the generic pointer to a __shared__ object on amdgcn).
__device__ __forceinline__ void async_cp16(const void* gptr, void* lptr) {
  unsigned lofs = (unsigned)(unsigned long long)lptr;
  unsigned long long ga = (unsigned long long)gptr;
  asm volatile("global_load_async_to_lds_b128 %0, %1, off"
               :: "v"(lofs), "v"(ga) : "memory");
}
__device__ __forceinline__ void async_wait0() {
  asm volatile("s_wait_asynccnt 0x0" ::: "memory");
}

// ------------------------- elementwise helpers -----------------------------

__global__ void k_f32_to_bf16(const float* __restrict__ src,
                              unsigned short* __restrict__ dst, int n) {
  int i = blockIdx.x * blockDim.x + threadIdx.x;
  if (i < n) dst[i] = f2bf(src[i]);
}

__global__ void k_gather_film(const int* __restrict__ mods,
                              const float* __restrict__ emb,
                              unsigned short* __restrict__ e_bf) {
  int i = blockIdx.x * blockDim.x + threadIdx.x;     // over MROWS*FILMD
  if (i >= MROWS * FILMD) return;
  int row = i / FILMD, j = i - row * FILMD;
  int m = mods[row]; if (m < 1) m = 1;
  e_bf[i] = f2bf(emb[(size_t)m * FILMD + j]);
}

__global__ void k_film(const float* __restrict__ g, const float* __restrict__ bfm,
                       const float* __restrict__ x, unsigned short* __restrict__ y_bf,
                       int n) {
  int i = blockIdx.x * blockDim.x + threadIdx.x;
  if (i < n) y_bf[i] = f2bf((1.0f + g[i]) * x[i] + bfm[i]);
}

// ---------------------- generic batched bf16 WMMA GEMM ---------------------
// C[M x Nc] = A[M x K] (row major, lda) @ B[K x Nc] (row major, ldb) + bias
// batch z: base += (z>>4)*s?b + (z&15)*s?h   (H=16 factorization; z=0 if unbatched)

__global__ __launch_bounds__(256) void k_gemm_bf16(
    const unsigned short* __restrict__ A, int lda, long long sAb, long long sAh,
    const unsigned short* __restrict__ Bm, int ldb, long long sBb, long long sBh,
    float* __restrict__ Cf, unsigned short* __restrict__ Cb, int ldc,
    long long sCb, long long sCh,
    const float* __restrict__ bias, int M, int Nc, int K, int relu) {
  __shared__ unsigned short sA[BM * LSA];
  __shared__ unsigned short sB[BN * LSB];     // transposed: sB[n][k]

  int z  = blockIdx.z;
  int bb = z >> 4, hh = z & 15;
  A  += bb * sAb + hh * sAh;
  Bm += bb * sBb + hh * sBh;
  long long cOff = bb * sCb + hh * sCh;
  if (Cf) Cf += cOff;
  if (Cb) Cb += cOff;

  int tid  = threadIdx.x;
  int lane = tid & 31;
  int wave = tid >> 5;
  int wm   = wave >> 1;                       // 0..3  (32 rows)
  int wn   = wave & 1;                        // 0..1  (64 cols)
  int rowBase = blockIdx.y * BM;
  int colBase = blockIdx.x * BN;

  v8f acc[2][4] = {};

  int arow  = wm * 32 + (lane & 15);
  int kbA   = (lane >> 4) * 8;                // A frag: K = kbA+{0..7}, kbA+16+{0..7}
  int bcol  = wn * 64 + (lane & 15);
  int kbB   = (lane >> 4) * 8;                // B frag: K = kbB+{0..7}, kbB+16+{0..7}

  for (int k0 = 0; k0 < K; k0 += BK) {
    // --- A tile (BM x BK): async global->LDS, 512 x b128, 2 per thread ---
#pragma unroll
    for (int t = 0; t < 2; t++) {
      int uidx = tid + t * 256;
      int r  = uidx >> 2;
      int kp = (uidx & 3) * 8;
      async_cp16(A + (size_t)(rowBase + r) * lda + k0 + kp, &sA[r * LSA + kp]);
    }
    // --- B tile (BK x BN) -> sB[n][k]: K-row pairs, packed b32 stores ---
    {
      int k2 = (tid >> 4) * 2;                // even k row 0..30
      int cg = (tid & 15) * 8;                // column group
      const unsigned short* bp = Bm + (size_t)(k0 + k2) * ldb + colBase + cg;
      uint4 v0 = *(const uint4*)bp;
      uint4 v1 = *(const uint4*)(bp + ldb);
      const unsigned short* p0 = (const unsigned short*)&v0;
      const unsigned short* p1 = (const unsigned short*)&v1;
#pragma unroll
      for (int e = 0; e < 8; e++) {
        unsigned w = (unsigned)p0[e] | ((unsigned)p1[e] << 16);
        *(unsigned*)&sB[(cg + e) * LSB + k2] = w;
      }
    }
    async_wait0();
    __syncthreads();

    Frag a[2], b[4];
#pragma unroll
    for (int mt = 0; mt < 2; mt++) {
      a[mt].q[0] = *(const uint4*)&sA[(arow + mt * 16) * LSA + kbA];
      a[mt].q[1] = *(const uint4*)&sA[(arow + mt * 16) * LSA + kbA + 16];
    }
#pragma unroll
    for (int nt = 0; nt < 4; nt++) {
      b[nt].q[0] = *(const uint4*)&sB[(bcol + nt * 16) * LSB + kbB];
      b[nt].q[1] = *(const uint4*)&sB[(bcol + nt * 16) * LSB + kbB + 16];
    }
#pragma unroll
    for (int mt = 0; mt < 2; mt++)
#pragma unroll
      for (int nt = 0; nt < 4; nt++)
        acc[mt][nt] = __builtin_amdgcn_wmma_f32_16x16x32_bf16(
            false, a[mt].v, false, b[nt].v, (short)0, acc[mt][nt], false, false);
    __syncthreads();
  }

  // --- epilogue: C layout lane n = lane&15, m = r + (lane>>4)*8 ---
  int cn = lane & 15;
  int mh = (lane >> 4) * 8;
#pragma unroll
  for (int mt = 0; mt < 2; mt++) {
#pragma unroll
    for (int nt = 0; nt < 4; nt++) {
      int gcol = colBase + wn * 64 + nt * 16 + cn;
      float bv = bias ? bias[gcol] : 0.0f;
      Acc8 t; t.v = acc[mt][nt];
#pragma unroll
      for (int r = 0; r < 8; r++) {
        int grow = rowBase + wm * 32 + mt * 16 + mh + r;
        float v = t.f[r] + bv;
        if (relu && v < 0.0f) v = 0.0f;
        size_t idx = (size_t)grow * ldc + gcol;
        if (Cf) Cf[idx] = v;
        if (Cb) Cb[idx] = f2bf(v);
      }
    }
  }
}

// ------------------- batched NT score GEMM: S = Q @ K^T --------------------
// per z = b*16+h: Q,K base b*Nseq*Dm + h*HD (rows stride Dm), S base z*Nseq*Nseq

__global__ __launch_bounds__(256) void k_scores(
    const unsigned short* __restrict__ Q, const unsigned short* __restrict__ Km,
    float* __restrict__ S, const float* __restrict__ mask, float scale) {
  __shared__ unsigned short sA[BM * LSA];
  __shared__ unsigned short sB[BN * LSB];     // sB[j][k] : row j of K-matrix

  int z  = blockIdx.z;
  int bb = z >> 4, hh = z & 15;
  const unsigned short* Ab = Q  + (size_t)bb * Nseq * Dm + hh * HD;
  const unsigned short* Bb = Km + (size_t)bb * Nseq * Dm + hh * HD;
  float* Sb = S + (size_t)z * Nseq * Nseq;

  int tid  = threadIdx.x;
  int lane = tid & 31;
  int wave = tid >> 5;
  int wm   = wave >> 1;
  int wn   = wave & 1;
  int rowBase = blockIdx.y * BM;
  int colBase = blockIdx.x * BN;

  v8f acc[2][4] = {};

  int arow = wm * 32 + (lane & 15);
  int kbA  = (lane >> 4) * 8;
  int bcol = wn * 64 + (lane & 15);
  int kbB  = (lane >> 4) * 8;

  for (int k0 = 0; k0 < HD; k0 += BK) {
    // both tiles are straight copies -> async global->LDS b128
#pragma unroll
    for (int t = 0; t < 2; t++) {
      int uidx = tid + t * 256;
      int r  = uidx >> 2;
      int kp = (uidx & 3) * 8;
      async_cp16(Ab + (size_t)(rowBase + r) * Dm + k0 + kp, &sA[r * LSA + kp]);
    }
#pragma unroll
    for (int t = 0; t < 2; t++) {
      int uidx = tid + t * 256;
      int r  = uidx >> 2;                     // 0..127 (K-matrix row j)
      int kp = (uidx & 3) * 8;
      async_cp16(Bb + (size_t)(colBase + r) * Dm + k0 + kp, &sB[r * LSB + kp]);
    }
    async_wait0();
    __syncthreads();

    Frag a[2], b[4];
#pragma unroll
    for (int mt = 0; mt < 2; mt++) {
      a[mt].q[0] = *(const uint4*)&sA[(arow + mt * 16) * LSA + kbA];
      a[mt].q[1] = *(const uint4*)&sA[(arow + mt * 16) * LSA + kbA + 16];
    }
#pragma unroll
    for (int nt = 0; nt < 4; nt++) {
      b[nt].q[0] = *(const uint4*)&sB[(bcol + nt * 16) * LSB + kbB];
      b[nt].q[1] = *(const uint4*)&sB[(bcol + nt * 16) * LSB + kbB + 16];
    }
#pragma unroll
    for (int mt = 0; mt < 2; mt++)
#pragma unroll
      for (int nt = 0; nt < 4; nt++)
        acc[mt][nt] = __builtin_amdgcn_wmma_f32_16x16x32_bf16(
            false, a[mt].v, false, b[nt].v, (short)0, acc[mt][nt], false, false);
    __syncthreads();
  }

  int cn = lane & 15;
  int mh = (lane >> 4) * 8;
#pragma unroll
  for (int mt = 0; mt < 2; mt++) {
#pragma unroll
    for (int nt = 0; nt < 4; nt++) {
      int gcol = colBase + wn * 64 + nt * 16 + cn;
      Acc8 t; t.v = acc[mt][nt];
#pragma unroll
      for (int r = 0; r < 8; r++) {
        int grow = rowBase + wm * 32 + mt * 16 + mh + r;
        Sb[(size_t)grow * Nseq + gcol] = t.f[r] * scale + mask[(size_t)grow * Nseq + gcol];
      }
    }
  }
}

// ------------------------- row softmax / entmax ----------------------------
// one wave per 512-element row; 8 rows per 256-thread block

__global__ __launch_bounds__(256) void k_softmax(const float* __restrict__ S,
                                                 unsigned short* __restrict__ P) {
  int row  = blockIdx.x * 8 + (threadIdx.x >> 5);
  int lane = threadIdx.x & 31;
  const float* src = S + (size_t)row * Nseq;
  float v[16];
  float mx = -3.402823466e38f;
#pragma unroll
  for (int i = 0; i < 16; i++) { v[i] = src[lane + i * 32]; mx = fmaxf(mx, v[i]); }
#pragma unroll
  for (int o = 16; o >= 1; o >>= 1) mx = fmaxf(mx, __shfl_xor(mx, o, 32));
  float s = 0.0f;
#pragma unroll
  for (int i = 0; i < 16; i++) { v[i] = __expf(v[i] - mx); s += v[i]; }
#pragma unroll
  for (int o = 16; o >= 1; o >>= 1) s += __shfl_xor(s, o, 32);
  float inv = 1.0f / s;
  unsigned short* dst = P + (size_t)row * Nseq;
#pragma unroll
  for (int i = 0; i < 16; i++) dst[lane + i * 32] = f2bf(v[i] * inv);
}

__global__ __launch_bounds__(256) void k_entmax(const float* __restrict__ S,
                                                unsigned short* __restrict__ P) {
  int row  = blockIdx.x * 8 + (threadIdx.x >> 5);
  int lane = threadIdx.x & 31;
  const float* src = S + (size_t)row * Nseq;
  float v[16];
  float mx = -3.402823466e38f;
#pragma unroll
  for (int i = 0; i < 16; i++) { v[i] = src[lane + i * 32]; mx = fmaxf(mx, v[i]); }
#pragma unroll
  for (int o = 16; o >= 1; o >>= 1) mx = fmaxf(mx, __shfl_xor(mx, o, 32));
#pragma unroll
  for (int i = 0; i < 16; i++) v[i] -= mx;     // max now 0

  const float am1 = 0.3f, inv_am1 = 1.0f / 0.3f;
  float lo = -10.0f, hi = 10.0f;
  for (int it = 0; it < 50; it++) {
    float mid = 0.5f * (lo + hi);
    float s = 0.0f;
#pragma unroll
    for (int i = 0; i < 16; i++) {
      float t = am1 * v[i] - mid;
      if (t > 0.0f) s += powf(t, inv_am1);
    }
#pragma unroll
    for (int o = 16; o >= 1; o >>= 1) s += __shfl_xor(s, o, 32);
    bool gt = s > 1.0f;
    lo = gt ? mid : lo;
    hi = gt ? hi : mid;
  }
  float p[16];
  float s = 0.0f;
#pragma unroll
  for (int i = 0; i < 16; i++) {
    float t = am1 * v[i] - hi;
    p[i] = (t > 0.0f) ? powf(t, inv_am1) : 0.0f;
    s += p[i];
  }
#pragma unroll
  for (int o = 16; o >= 1; o >>= 1) s += __shfl_xor(s, o, 32);
  float inv = 1.0f / (s + 1e-12f);
  unsigned short* dst = P + (size_t)row * Nseq;
#pragma unroll
  for (int i = 0; i < 16; i++) dst[lane + i * 32] = f2bf(p[i] * inv);
}

// ----------------------- gate + residual + layernorm -----------------------

__global__ __launch_bounds__(256) void k_gate_ln1(
    const float* __restrict__ x, const float* __restrict__ zi,
    const float* __restrict__ ze, const float* __restrict__ ga,
    const float* __restrict__ g, const float* __restrict__ b,
    float* __restrict__ h, unsigned short* __restrict__ hbf) {
  __shared__ float red[256];
  int row = blockIdx.x, tid = threadIdx.x;
  float a0 = ga[0], a1 = ga[1];
  float am = fmaxf(a0, a1);
  float e0 = __expf(a0 - am), e1 = __expf(a1 - am);
  float w0 = e0 / (e0 + e1), w1 = e1 / (e0 + e1);
  size_t base = (size_t)row * Dm;
  float t[8]; float s = 0.0f;
#pragma unroll
  for (int i = 0; i < 8; i++) {
    int c = tid + i * 256;
    t[i] = x[base + c] + w0 * zi[base + c] + w1 * ze[base + c];
    s += t[i];
  }
  red[tid] = s; __syncthreads();
  for (int o = 128; o >= 1; o >>= 1) { if (tid < o) red[tid] += red[tid + o]; __syncthreads(); }
  float mean = red[0] * (1.0f / Dm);
  __syncthreads();
  float vs = 0.0f;
#pragma unroll
  for (int i = 0; i < 8; i++) { float d = t[i] - mean; vs += d * d; }
  red[tid] = vs; __syncthreads();
  for (int o = 128; o >= 1; o >>= 1) { if (tid < o) red[tid] += red[tid + o]; __syncthreads(); }
  float inv = rsqrtf(red[0] * (1.0f / Dm) + 1e-5f);
#pragma unroll
  for (int i = 0; i < 8; i++) {
    int c = tid + i * 256;
    float hv = (t[i] - mean) * inv * g[c] + b[c];
    h[base + c] = hv;
    hbf[base + c] = f2bf(hv);
  }
}

__global__ __launch_bounds__(256) void k_ln2(
    const float* __restrict__ h, const float* __restrict__ y,
    const float* __restrict__ g, const float* __restrict__ b,
    float* __restrict__ out) {
  __shared__ float red[256];
  int row = blockIdx.x, tid = threadIdx.x;
  size_t base = (size_t)row * Dm;
  float t[8]; float s = 0.0f;
#pragma unroll
  for (int i = 0; i < 8; i++) { int c = tid + i * 256; t[i] = h[base + c] + y[base + c]; s += t[i]; }
  red[tid] = s; __syncthreads();
  for (int o = 128; o >= 1; o >>= 1) { if (tid < o) red[tid] += red[tid + o]; __syncthreads(); }
  float mean = red[0] * (1.0f / Dm);
  __syncthreads();
  float vs = 0.0f;
#pragma unroll
  for (int i = 0; i < 8; i++) { float d = t[i] - mean; vs += d * d; }
  red[tid] = vs; __syncthreads();
  for (int o = 128; o >= 1; o >>= 1) { if (tid < o) red[tid] += red[tid + o]; __syncthreads(); }
  float inv = rsqrtf(red[0] * (1.0f / Dm) + 1e-5f);
#pragma unroll
  for (int i = 0; i < 8; i++) {
    int c = tid + i * 256;
    out[base + c] = (t[i] - mean) * inv * g[c] + b[c];
  }
}

// ---------------------------------------------------------------------------

extern "C" void kernel_launch(void* const* d_in, const int* in_sizes, int n_in,
                              void* d_out, int out_size, void* d_ws, size_t ws_size,
                              hipStream_t stream) {
  (void)in_sizes; (void)n_in; (void)out_size; (void)ws_size;

  const float* x        = (const float*)d_in[0];
  const float* mask_a   = (const float*)d_in[1];
  const float* mask_e   = (const float*)d_in[2];
  const int*   mods     = (const int*)d_in[3];
  const float* film_emb = (const float*)d_in[4];
  const float* film_gw  = (const float*)d_in[5];
  const float* film_gb  = (const float*)d_in[6];
  const float* film_bw  = (const float*)d_in[7];
  const float* film_bb  = (const float*)d_in[8];
  const float* gate_a   = (const float*)d_in[9];
  const float* ln1_g    = (const float*)d_in[10];
  const float* ln1_b    = (const float*)d_in[11];
  const float* ln2_g    = (const float*)d_in[12];
  const float* ln2_b    = (const float*)d_in[13];
  const float* ffn_w1   = (const float*)d_in[14];
  const float* ffn_b1   = (const float*)d_in[15];
  const float* ffn_w2   = (const float*)d_in[16];
  const float* ffn_b2   = (const float*)d_in[17];
  // 18..25: wq_a,bq_a,wk_a,bk_a,wv_a,bv_a,wo_a,bo_a ; 26..33: same for 'e'

  // ----- workspace layout -----
  char* ws = (char*)d_ws;
  size_t off = 0;
  auto take = [&](size_t bytes) -> char* {
    char* p = ws + off;
    off += (bytes + 255) & ~(size_t)255;
    return p;
  };
  unsigned short* wbf[8];
  for (int i = 0; i < 8; i++) wbf[i] = (unsigned short*)take((size_t)Dm * Dm * 2);
  unsigned short* w1bf  = (unsigned short*)take((size_t)Dm * Ff * 2);
  unsigned short* w2bf  = (unsigned short*)take((size_t)Ff * Dm * 2);
  unsigned short* fgwbf = (unsigned short*)take((size_t)FILMD * Dm * 2);
  unsigned short* fbwbf = (unsigned short*)take((size_t)FILMD * Dm * 2);
  unsigned short* ebf   = (unsigned short*)take((size_t)MROWS * FILMD * 2);
  unsigned short* xbf   = (unsigned short*)take((size_t)MROWS * Dm * 2);
  unsigned short* ybf   = (unsigned short*)take((size_t)MROWS * Dm * 2);
  float*          hf    = (float*)take((size_t)MROWS * Dm * 4);
  unsigned short* hbf   = (unsigned short*)take((size_t)MROWS * Dm * 2);
  float*          zi    = (float*)take((size_t)MROWS * Dm * 4);
  float*          ze    = (float*)take((size_t)MROWS * Dm * 4);
  char*           tmp   = take((size_t)128 * 1024 * 1024);   // phase scratch
  // phase views inside tmp
  float*          gf    = (float*)tmp;                                   // 16 MiB
  float*          bfm   = (float*)(tmp + (size_t)MROWS * Dm * 4);        // 16 MiB
  unsigned short* qbf   = (unsigned short*)tmp;                          // 8 MiB
  unsigned short* kbf   = (unsigned short*)(tmp + (size_t)MROWS * Dm * 2);
  unsigned short* vbf   = (unsigned short*)(tmp + (size_t)MROWS * Dm * 4);
  char*           atn   = tmp + (size_t)MROWS * Dm * 6;                  // +24 MiB
  float*          sbuf  = (float*)atn;                                   // 64 MiB
  unsigned short* pbf   = (unsigned short*)(atn + (size_t)Bsz * NH * Nseq * Nseq * 4);
  unsigned short* obf   = (unsigned short*)(atn + (size_t)Bsz * NH * Nseq * Nseq * 6);
  unsigned short* midbf = (unsigned short*)tmp;                          // 32 MiB
  float*          yf    = (float*)(tmp + (size_t)MROWS * Ff * 2);        // 16 MiB

  auto cvt = [&](const float* s, unsigned short* d, size_t n) {
    k_f32_to_bf16<<<(unsigned)((n + 255) / 256), 256, 0, stream>>>(s, d, (int)n);
  };
  auto gemm = [&](const unsigned short* A, int lda, long long sAb, long long sAh,
                  const unsigned short* Bm, int ldb, long long sBb, long long sBh,
                  float* Cf, unsigned short* Cb, int ldc, long long sCb, long long sCh,
                  const float* bias, int M, int Nc, int K, int relu, int batches) {
    dim3 g(Nc / BN, M / BM, batches);
    k_gemm_bf16<<<g, 256, 0, stream>>>(A, lda, sAb, sAh, Bm, ldb, sBb, sBh,
                                       Cf, Cb, ldc, sCb, sCh, bias, M, Nc, K, relu);
  };

  // ----- weight / activation conversion (deterministic every call) -----
  static const int widx[8] = {18, 20, 22, 24, 26, 28, 30, 32};
  for (int i = 0; i < 8; i++) cvt((const float*)d_in[widx[i]], wbf[i], (size_t)Dm * Dm);
  cvt(ffn_w1, w1bf, (size_t)Dm * Ff);
  cvt(ffn_w2, w2bf, (size_t)Ff * Dm);
  cvt(film_gw, fgwbf, (size_t)FILMD * Dm);
  cvt(film_bw, fbwbf, (size_t)FILMD * Dm);
  cvt(x, xbf, (size_t)MROWS * Dm);

  // ----- FiLM: y_intra = (1 + e@gw + gb) * x + (e@bw + bb) -----
  k_gather_film<<<(MROWS * FILMD) / 256, 256, 0, stream>>>(mods, film_emb, ebf);
  gemm(ebf, FILMD, 0, 0, fgwbf, Dm, 0, 0, gf,  nullptr, Dm, 0, 0, film_gb, MROWS, Dm, FILMD, 0, 1);
  gemm(ebf, FILMD, 0, 0, fbwbf, Dm, 0, 0, bfm, nullptr, Dm, 0, 0, film_bb, MROWS, Dm, FILMD, 0, 1);
  k_film<<<(MROWS * Dm) / 256, 256, 0, stream>>>(gf, bfm, x, ybf, MROWS * Dm);

  // ----- two attention branches -----
  const float scale = 0.08838834764831845f;   // 1/sqrt(128)
  for (int br = 0; br < 2; br++) {
    const unsigned short* inp = (br == 0) ? ybf : xbf;
    int ib = 18 + br * 8;
    unsigned short* WQ = wbf[br * 4 + 0];
    unsigned short* WK = wbf[br * 4 + 1];
    unsigned short* WV = wbf[br * 4 + 2];
    unsigned short* WO = wbf[br * 4 + 3];
    // projections (bf16 out only)
    gemm(inp, Dm, 0, 0, WQ, Dm, 0, 0, nullptr, qbf, Dm, 0, 0, (const float*)d_in[ib + 1], MROWS, Dm, Dm, 0, 1);
    gemm(inp, Dm, 0, 0, WK, Dm, 0, 0, nullptr, kbf, Dm, 0, 0, (const float*)d_in[ib + 3], MROWS, Dm, Dm, 0, 1);
    gemm(inp, Dm, 0, 0, WV, Dm, 0, 0, nullptr, vbf, Dm, 0, 0, (const float*)d_in[ib + 5], MROWS, Dm, Dm, 0, 1);
    // scores: (B*H) x N x N, scaled + mask
    dim3 gs(Nseq / BN, Nseq / BM, Bsz * NH);
    k_scores<<<gs, 256, 0, stream>>>(qbf, kbf, sbuf, (br == 0) ? mask_a : mask_e, scale);
    // probabilities
    if (br == 0) k_softmax<<<(Bsz * NH * Nseq) / 8, 256, 0, stream>>>(sbuf, pbf);
    else         k_entmax <<<(Bsz * NH * Nseq) / 8, 256, 0, stream>>>(sbuf, pbf);
    // A@V (batched, head-strided output into (B,N,D) layout)
    gemm(pbf, Nseq, (long long)NH * Nseq * Nseq, (long long)Nseq * Nseq,
         vbf, Dm, (long long)Nseq * Dm, (long long)HD,
         nullptr, obf, Dm, (long long)Nseq * Dm, (long long)HD,
         nullptr, Nseq, HD, Nseq, 0, Bsz * NH);
    // output projection
    gemm(obf, Dm, 0, 0, WO, Dm, 0, 0, (br == 0) ? zi : ze, nullptr, Dm, 0, 0,
         (const float*)d_in[ib + 7], MROWS, Dm, Dm, 0, 1);
  }

  // ----- gate + residual + LN1, FFN, LN2 -----
  k_gate_ln1<<<MROWS, 256, 0, stream>>>(x, zi, ze, gate_a, ln1_g, ln1_b, hf, hbf);
  gemm(hbf,   Dm, 0, 0, w1bf, Ff, 0, 0, nullptr, midbf, Ff, 0, 0, ffn_b1, MROWS, Ff, Dm, 1, 1);
  gemm(midbf, Ff, 0, 0, w2bf, Dm, 0, 0, yf,     nullptr, Dm, 0, 0, ffn_b2, MROWS, Dm, Ff, 0, 1);
  k_ln2<<<MROWS, 256, 0, stream>>>(hf, yf, ln2_g, ln2_b, (float*)d_out);
}
